// MyLSTM_58463094833753
// MI455X (gfx1250) — compile-verified, matrix operand order checked by reading, not run on hardware
//
#include <hip/hip_runtime.h>
#include <hip/hip_bf16.h>

typedef __attribute__((ext_vector_type(16))) _Float16 v16h;
typedef __attribute__((ext_vector_type(8)))  float    v8f;
typedef __attribute__((ext_vector_type(4)))  int      v4i;
typedef __attribute__((ext_vector_type(4)))  unsigned int v4u;
typedef __attribute__((ext_vector_type(8)))  int      v8i;

constexpr int T = 1024, B = 64, I = 1024, H = 1024;
constexpr int TB = T * B;       // 65536 rows of the x-projection GEMM
constexpr int BH = B * H;       // 65536 elements of carry state
constexpr int LDSPAD = 40;      // row stride (halfs), mult of 8 for b128 alignment

// ---------------- workspace layout (bytes) ----------------
constexpr size_t WS_INP_H  = 0;                                        // [TB][I] f16   (128 MB)
constexpr size_t WS_WT     = WS_INP_H + (size_t)TB * I * 2;            // 8 slabs [N][K] f16 (16 MB)
constexpr size_t WS_XPROJ  = WS_WT    + (size_t)8 * H * I * 2;         // 4 slabs [TB][H] f16 (512 MB)
constexpr size_t WS_HPROJ  = WS_XPROJ + (size_t)4 * TB * H * 2;        // 4 slabs [B][H] f32 (1 MB)
constexpr size_t WS_STATEH = WS_HPROJ + (size_t)4 * BH * 4;            // [B][H] f16 (128 KB)

struct Ptrs8  { const float* p[8]; };
struct Bias4  { const float* b[4]; };

// ---------------- CDNA5 async global->LDS path (probe-verified) ----------------
#if defined(__has_builtin)
#if __has_builtin(__builtin_amdgcn_global_load_async_to_lds_b128) && \
    __has_builtin(__builtin_amdgcn_s_wait_asynccnt)
#define HAVE_ASYNC_LDS 1
#endif
#if __has_builtin(__builtin_amdgcn_tensor_load_to_lds) && \
    __has_builtin(__builtin_amdgcn_s_wait_tensorcnt)
#define HAVE_TDM 1
#endif
#endif
#ifndef HAVE_ASYNC_LDS
#define HAVE_ASYNC_LDS 0
#endif
#ifndef HAVE_TDM
#define HAVE_TDM 0
#endif

// 16B global -> LDS copy; async (no VGPR round-trip, ASYNCcnt) when available.
__device__ __forceinline__ void copy_b128(const _Float16* g, _Float16* l) {
#if HAVE_ASYNC_LDS
    __builtin_amdgcn_global_load_async_to_lds_b128(
        (__attribute__((address_space(1))) v4i*)g,
        (__attribute__((address_space(3))) v4i*)l, 0, 0);
#else
    *reinterpret_cast<uint4*>(l) = *reinterpret_cast<const uint4*>(g);
#endif
}

#if HAVE_TDM
// TDM: DMA a 128x32-half 2D tile (rows stride 1024 halfs) from global into LDS,
// inserting 16B of pad every 64B so LDS rows land at the 40-half padded stride.
// Issued by one wave per workgroup; tracked by TENSORcnt.
// This toolchain exposes the 6-arg builtin: (v4u g0, v8i g1, v4i g2, v4i g3, v8i g4, i32 cpol).
__device__ __forceinline__ void tdm_load_tile_128x32(const _Float16* gsrc, unsigned lds_byte_off) {
    const unsigned long long ga = (unsigned long long)(size_t)gsrc;
    v4u g0;
    g0.x = 1u;                                                    // count=1 (valid user D#)
    g0.y = lds_byte_off;                                          // lds_addr (bytes)
    g0.z = (unsigned)(ga & 0xFFFFFFFFu);                          // global_addr[31:0]
    g0.w = (unsigned)((ga >> 32) & 0x01FFFFFFu) | 0x80000000u;    // global_addr[56:32], type=2
    v8i g1;
    g1[0] = (int)((1u << 16) |        // data_size = 1 -> 2 bytes
                  (1u << 20) |        // pad_enable
                  (3u << 22) |        // pad_interval = 16 DWORDs (64 B)
                  (3u << 25));        // pad_amount   = 4 DWORDs (16 B)
    g1[1] = (int)(1024u << 16);       // tensor_dim0 = 1024 (bits 79:48, low part)
    g1[2] = (int)(1024u << 16);       // tensor_dim0 hi=0 ; tensor_dim1 = 1024 (low part)
    g1[3] = (int)(32u << 16);         // tensor_dim1 hi=0 ; tile_dim0 = 32
    g1[4] = 128;                      // tile_dim1 = 128 ; tile_dim2 = 0
    g1[5] = 1024;                     // tensor_dim0_stride = 1024 elements (low 32)
    g1[6] = 0;                        // stride hi ; tensor_dim1_stride (unused, 2D)
    g1[7] = 0;
    const v4i z4 = {0, 0, 0, 0};
    const v8i z8 = {0, 0, 0, 0, 0, 0, 0, 0};
    __builtin_amdgcn_tensor_load_to_lds(g0, g1, z4, z4, z8, 0);
}
#endif

// ---------------- fragment loader ----------------
// Per-lane layout for A (16x32, MxK) and B' (16x32, NxK from transposed weights):
// lane L (L%16 = row), halfs 0..7 = K kb..kb+7, halfs 8..15 = K kb+16..kb+23,
// kb = (L>=16)*8.  Two ds_load_b128 per fragment.
__device__ __forceinline__ v16h load_frag16(const _Float16* lds, int row0, int rowStride) {
    const int lane = threadIdx.x & 31;
    const _Float16* p = lds + (size_t)(row0 + (lane & 15)) * rowStride + ((lane >> 4) << 3);
    v16h f;
    reinterpret_cast<uint4*>(&f)[0] = *reinterpret_cast<const uint4*>(p);
    reinterpret_cast<uint4*>(&f)[1] = *reinterpret_cast<const uint4*>(p + 16);
    return f;
}

// ---------------- conversion kernels ----------------
__global__ void k_cvt_f32_to_f16(const float* __restrict__ in, _Float16* __restrict__ out, int n4) {
    int i = blockIdx.x * blockDim.x + threadIdx.x;
    if (i >= n4) return;
    float4 v = reinterpret_cast<const float4*>(in)[i];
    union { _Float16 h[4]; uint2 u; } r;
    r.h[0] = (_Float16)v.x; r.h[1] = (_Float16)v.y;
    r.h[2] = (_Float16)v.z; r.h[3] = (_Float16)v.w;
    reinterpret_cast<uint2*>(out)[i] = r.u;
}

// W[k][n] (f32, row-major [1024][1024]) -> O[n][k] (f16) ; coalesced writes
__global__ void k_transpose_cvt(Ptrs8 wp, _Float16* __restrict__ out) {
    const int m = blockIdx.z;
    const float* __restrict__ W = wp.p[m];
    _Float16* __restrict__ O = out + (size_t)m * H * I;
    int id = blockIdx.x * blockDim.x + threadIdx.x;   // 0 .. H*I-1
    int n = id >> 10;
    int k = id & 1023;
    O[id] = (_Float16)W[(size_t)k * H + n];
}

// ---------------- phase 1: x-projections, big WMMA GEMM ----------------
// grid (TB/128, H/128, 4 gates), block 256 (8 waves), 128x128 tile, K step 32,
// double-buffered LDS with async global->LDS staging.
__global__ __launch_bounds__(256)
void k_xproj_wmma(const _Float16* __restrict__ Ah,   // [TB][I]
                  const _Float16* __restrict__ Wt,   // 4 slabs [H][I] (transposed weights)
                  Bias4 bias,
                  _Float16* __restrict__ xproj)      // 4 slabs [TB][H]
{
    constexpr int ABUF = 128 * LDSPAD;
    constexpr int NK   = I / 32;
    __shared__ __align__(16) _Float16 As[2 * ABUF];
    __shared__ __align__(16) _Float16 Bs[2 * ABUF];

    const int g    = blockIdx.z;
    const int row0 = blockIdx.x * 128;
    const int col0 = blockIdx.y * 128;
    const int tid  = threadIdx.x;
    const int wave = tid >> 5;
    const int lane = tid & 31;
    const int mq   = (wave & 1) * 64;   // wave tile: 64 rows x 32 cols
    const int nq   = (wave >> 1) * 32;

    const _Float16* __restrict__ Wg = Wt + (size_t)g * H * I;

    v8f acc[4][2];
    const v8f vzero = {0.f,0.f,0.f,0.f,0.f,0.f,0.f,0.f};
#pragma unroll
    for (int mt = 0; mt < 4; ++mt)
#pragma unroll
        for (int nt = 0; nt < 2; ++nt) acc[mt][nt] = vzero;

    const int arow = tid >> 1;               // 0..127
    const int aseg = (tid & 1) * 16;         // 16 halfs (32 B) per thread per matrix
    const _Float16* agp = Ah + (size_t)(row0 + arow) * I + aseg;
    const _Float16* bgp = Wg + (size_t)(col0 + arow) * I + aseg;
    _Float16* asp = As + arow * LDSPAD + aseg;
    _Float16* bsp = Bs + arow * LDSPAD + aseg;

    // prologue: stage tile 0 into buffer 0
    copy_b128(agp,     asp);
    copy_b128(agp + 8, asp + 8);
    copy_b128(bgp,     bsp);
    copy_b128(bgp + 8, bsp + 8);

    for (int kt = 0; kt < NK; ++kt) {
        const int cur = kt & 1;
        if (kt + 1 < NK) {
            const int k1 = (kt + 1) * 32;
            const int nxt = (1 - cur) * ABUF;
            copy_b128(agp + k1,     asp + nxt);
            copy_b128(agp + k1 + 8, asp + nxt + 8);
            copy_b128(bgp + k1,     bsp + nxt);
            copy_b128(bgp + k1 + 8, bsp + nxt + 8);
#if HAVE_ASYNC_LDS
            __builtin_amdgcn_s_wait_asynccnt(4);
#endif
        } else {
#if HAVE_ASYNC_LDS
            __builtin_amdgcn_s_wait_asynccnt(0);
#endif
        }
        __syncthreads();

        const _Float16* Ac = As + cur * ABUF;
        const _Float16* Bc = Bs + cur * ABUF;
        v16h af[4], bf[2];
#pragma unroll
        for (int mt = 0; mt < 4; ++mt) af[mt] = load_frag16(Ac, mq + mt * 16, LDSPAD);
#pragma unroll
        for (int nt = 0; nt < 2; ++nt) bf[nt] = load_frag16(Bc, nq + nt * 16, LDSPAD);
#pragma unroll
        for (int mt = 0; mt < 4; ++mt)
#pragma unroll
            for (int nt = 0; nt < 2; ++nt)
                acc[mt][nt] = __builtin_amdgcn_wmma_f32_16x16x32_f16(
                    false, af[mt], false, bf[nt], (short)0, acc[mt][nt], false, false);
        __syncthreads();
    }

    // C/D layout: vgpr v, lanes 0-15 -> M=v, lanes 16-31 -> M=8+v ; N = lane%16
    const int n   = lane & 15;
    const int mhi = (lane >> 4) * 8;
    const float* __restrict__ bptr = bias.b[g];
    _Float16* __restrict__ outg = xproj + (size_t)g * TB * H;
#pragma unroll
    for (int mt = 0; mt < 4; ++mt) {
#pragma unroll
        for (int nt = 0; nt < 2; ++nt) {
            const int col = col0 + nq + nt * 16 + n;
            const float bv = bptr[col];
#pragma unroll
            for (int v = 0; v < 8; ++v) {
                const int row = row0 + mq + mt * 16 + mhi + v;
                outg[(size_t)row * H + col] = (_Float16)(acc[mt][nt][v] + bv);
            }
        }
    }
}

// ---------------- phase 2a: per-step hidden projection GEMM ----------------
// grid (H/128, 4 gates), block 256 (8 waves). M=64, N=128 tile, K=1024.
// A tile via async global->LDS; B (weight) tile via TDM tensor_load_to_lds
// (one DMA per K-tile, wave 0, TENSORcnt) with descriptor-driven LDS padding.
__global__ __launch_bounds__(256)
void k_hproj_wmma(const _Float16* __restrict__ Sh,    // [B][H] state (f16)
                  const _Float16* __restrict__ Wht,   // 4 slabs [H][H] (transposed)
                  float* __restrict__ hproj)          // 4 slabs [B][H] f32
{
    constexpr int ABUF = 64  * LDSPAD;
    constexpr int BBUF = 128 * LDSPAD;
    constexpr int NK   = H / 32;
    __shared__ __align__(16) _Float16 As[2 * ABUF];
    __shared__ __align__(16) _Float16 Bs[2 * BBUF];

    const int g    = blockIdx.y;
    const int col0 = blockIdx.x * 128;
    const int tid  = threadIdx.x;
    const int wave = tid >> 5;
    const int lane = tid & 31;
    const int mq   = (wave & 1) * 32;   // wave tile: 32 rows x 32 cols
    const int nq   = (wave >> 1) * 32;

    const _Float16* __restrict__ Wg = Wht + (size_t)g * H * H;

    v8f acc[2][2];
    const v8f vzero = {0.f,0.f,0.f,0.f,0.f,0.f,0.f,0.f};
#pragma unroll
    for (int mt = 0; mt < 2; ++mt)
#pragma unroll
        for (int nt = 0; nt < 2; ++nt) acc[mt][nt] = vzero;

    const int arow = tid >> 2;              // 0..63 (A: 64x32, one 16B chunk / thread)
    const int aseg = (tid & 3) * 8;
    const _Float16* agp = Sh + (size_t)arow * H + aseg;
    _Float16* asp = As + arow * LDSPAD + aseg;

#if HAVE_TDM
    const _Float16* btile = Wg + (size_t)col0 * H;     // B tile origin (rows=cols of W', stride H)
    const unsigned ldsB = (unsigned)(size_t)((__attribute__((address_space(3))) void*)&Bs[0]);
    // prologue: A tile 0 (async) + B tile 0 (TDM)
    copy_b128(agp, asp);
    if (wave == 0) tdm_load_tile_128x32(btile, ldsB);
#else
    const int brow = tid >> 1;              // 0..127 (B: 128x32, two 16B chunks / thread)
    const int bseg = (tid & 1) * 16;
    const _Float16* bgp = Wg + (size_t)(col0 + brow) * H + bseg;
    _Float16* bsp = Bs + brow * LDSPAD + bseg;
    copy_b128(agp,     asp);
    copy_b128(bgp,     bsp);
    copy_b128(bgp + 8, bsp + 8);
#endif

    for (int kt = 0; kt < NK; ++kt) {
        const int cur = kt & 1;
        if (kt + 1 < NK) {
            const int k1 = (kt + 1) * 32;
#if HAVE_TDM
            copy_b128(agp + k1, asp + (1 - cur) * ABUF);
            if (wave == 0) {
                tdm_load_tile_128x32(btile + k1, ldsB + (unsigned)((1 - cur) * BBUF * 2));
                __builtin_amdgcn_s_wait_tensorcnt(1);
            }
#if HAVE_ASYNC_LDS
            __builtin_amdgcn_s_wait_asynccnt(1);
#endif
#else
            copy_b128(agp + k1,     asp + (1 - cur) * ABUF);
            copy_b128(bgp + k1,     bsp + (1 - cur) * BBUF);
            copy_b128(bgp + k1 + 8, bsp + (1 - cur) * BBUF + 8);
#if HAVE_ASYNC_LDS
            __builtin_amdgcn_s_wait_asynccnt(3);
#endif
#endif
        } else {
#if HAVE_TDM
            if (wave == 0) __builtin_amdgcn_s_wait_tensorcnt(0);
#endif
#if HAVE_ASYNC_LDS
            __builtin_amdgcn_s_wait_asynccnt(0);
#endif
        }
        __syncthreads();

        const _Float16* Ac = As + cur * ABUF;
        const _Float16* Bc = Bs + cur * BBUF;
        v16h af[2], bf[2];
#pragma unroll
        for (int mt = 0; mt < 2; ++mt) af[mt] = load_frag16(Ac, mq + mt * 16, LDSPAD);
#pragma unroll
        for (int nt = 0; nt < 2; ++nt) bf[nt] = load_frag16(Bc, nq + nt * 16, LDSPAD);
#pragma unroll
        for (int mt = 0; mt < 2; ++mt)
#pragma unroll
            for (int nt = 0; nt < 2; ++nt)
                acc[mt][nt] = __builtin_amdgcn_wmma_f32_16x16x32_f16(
                    false, af[mt], false, bf[nt], (short)0, acc[mt][nt], false, false);
        __syncthreads();
    }

    const int n   = lane & 15;
    const int mhi = (lane >> 4) * 8;
    float* __restrict__ outg = hproj + (size_t)g * BH;
#pragma unroll
    for (int mt = 0; mt < 2; ++mt) {
#pragma unroll
        for (int nt = 0; nt < 2; ++nt) {
            const int col = col0 + nq + nt * 16 + n;
#pragma unroll
            for (int v = 0; v < 8; ++v) {
                const int row = mq + mt * 16 + mhi + v;
                outg[(size_t)row * H + col] = acc[mt][nt][v];
            }
        }
    }
}

// ---------------- phase 2b: fused gate/update elementwise ----------------
__global__ void k_lstm_step(const _Float16* __restrict__ xproj,  // 4 slabs [TB][H] f16
                            const float* __restrict__ hproj,     // 4 slabs [B][H] f32
                            float* __restrict__ states,          // d_out [T][B][H]
                            float* __restrict__ memv,            // d_out tail (memory)
                            float* __restrict__ statev,          // d_out tail (state)
                            _Float16* __restrict__ state_h,      // f16 state for next GEMM
                            int t)
{
    const int idx = blockIdx.x * blockDim.x + threadIdx.x;   // < BH
    const size_t toff = (size_t)t * BH + idx;
    const size_t slab = (size_t)TB * H;
    const float xf = (float)xproj[0 * slab + toff];
    const float xi = (float)xproj[1 * slab + toff];
    const float xo = (float)xproj[2 * slab + toff];
    const float xn = (float)xproj[3 * slab + toff];
    const float f = 1.f / (1.f + __expf(-(xf + hproj[0 * BH + idx])));
    const float i = 1.f / (1.f + __expf(-(xi + hproj[1 * BH + idx])));
    const float o = 1.f / (1.f + __expf(-(xo + hproj[2 * BH + idx])));
    const float n = tanhf(xn + hproj[3 * BH + idx]);
    const float mem = f * memv[idx] + i * n;
    const float st  = o * tanhf(mem);
    memv[idx]   = mem;
    statev[idx] = st;
    states[toff] = st;
    state_h[idx] = (_Float16)st;
}

__global__ void k_init(const float* __restrict__ m0, const float* __restrict__ s0,
                       float* __restrict__ memv, float* __restrict__ statev,
                       _Float16* __restrict__ state_h)
{
    const int idx = blockIdx.x * blockDim.x + threadIdx.x;
    memv[idx] = m0[idx];
    const float s = s0[idx];
    statev[idx] = s;
    state_h[idx] = (_Float16)s;
}

// ---------------- launch ----------------
extern "C" void kernel_launch(void* const* d_in, const int* in_sizes, int n_in,
                              void* d_out, int out_size, void* d_ws, size_t ws_size,
                              hipStream_t stream) {
    const float* input   = (const float*)d_in[0];
    const float* memory0 = (const float*)d_in[1];
    const float* state0  = (const float*)d_in[2];

    char* ws = (char*)d_ws;
    _Float16* inp_h   = (_Float16*)(ws + WS_INP_H);
    _Float16* w_t     = (_Float16*)(ws + WS_WT);      // slabs 0..3 = Wx(f,i,o,n); 4..7 = Wh
    _Float16* xproj_h = (_Float16*)(ws + WS_XPROJ);
    float*    hproj   = (float*)(ws + WS_HPROJ);
    _Float16* state_h = (_Float16*)(ws + WS_STATEH);

    float* out    = (float*)d_out;
    float* states = out;                      // [T][B][H]
    float* memv   = out + (size_t)TB * H;     // memory
    float* statev = memv + BH;                // state

    // 1) input f32 -> f16
    const int n4 = TB * I / 4;
    k_cvt_f32_to_f16<<<n4 / 256, 256, 0, stream>>>(input, inp_h, n4);

    // 2) weights: transpose + convert to f16 slabs [N][K]
    Ptrs8 wp;
    wp.p[0] = (const float*)d_in[3];   // weight_xf
    wp.p[1] = (const float*)d_in[6];   // weight_xi
    wp.p[2] = (const float*)d_in[9];   // weight_xo
    wp.p[3] = (const float*)d_in[12];  // weight_xn
    wp.p[4] = (const float*)d_in[4];   // weight_hf
    wp.p[5] = (const float*)d_in[7];   // weight_hi
    wp.p[6] = (const float*)d_in[10];  // weight_ho
    wp.p[7] = (const float*)d_in[13];  // weight_hn
    k_transpose_cvt<<<dim3(H * I / 256, 1, 8), 256, 0, stream>>>(wp, w_t);

    // 3) big WMMA GEMM: x-projections for all 4 gates (+bias)
    Bias4 bp;
    bp.b[0] = (const float*)d_in[5];
    bp.b[1] = (const float*)d_in[8];
    bp.b[2] = (const float*)d_in[11];
    bp.b[3] = (const float*)d_in[14];
    k_xproj_wmma<<<dim3(TB / 128, H / 128, 4), 256, 0, stream>>>(inp_h, w_t, bp, xproj_h);

    // 4) init carry (memory/state live directly in d_out tail)
    k_init<<<BH / 256, 256, 0, stream>>>(memory0, state0, memv, statev, state_h);

    // 5) sequential recurrence: GEMM + fused elementwise per step
    const _Float16* wh_t = w_t + (size_t)4 * H * I;
    for (int t = 0; t < T; ++t) {
        k_hproj_wmma<<<dim3(H / 128, 4), 256, 0, stream>>>(state_h, wh_t, hproj);
        k_lstm_step<<<BH / 256, 256, 0, stream>>>(xproj_h, hproj, states, memv, statev, state_h, t);
    }
}
